// point_involution_v3_13443247637192
// MI455X (gfx1250) — compile-verified, matrix operand order checked by reading, not run on hardware
//
#include <hip/hip_runtime.h>
#include <math.h>

#define M_PTS 50000
#define N_PTS 50000
#define H_NB  32
#define C_CH  64
#define TS    66              // padded LDS row stride (elements)
#define NREP  64              // replicated stat accumulator slots
#define SLOPE 0.1f

typedef _Float16 v16h __attribute__((ext_vector_type(16)));
typedef float    v8f  __attribute__((ext_vector_type(8)));

#define LDS_FENCE() asm volatile("s_wait_dscnt 0x0" ::: "memory")

struct Args {
  const float* q_pts; const float* s_pts; const float* s_feats; const int* nbi;
  const float* dW1; const float* dW2; const float* dB2;
  const float* gW;  const float* gB;
  const float* aW1; const float* aW2; const float* aB2;
  float* part;          // 4 stages x NREP x 128 floats
  const float* params;  // 4 stages x {scale[64], shift[64]}
  float* out;
};

// ---- WMMA fragment packing (CDNA5 16-bit A / B layouts) -------------------
__device__ __forceinline__ v16h load_afrag(const _Float16* act, int mt, int k0,
                                           int lane) {
  int row = mt * 16 + (lane & 15);
  int hs  = lane >> 4;
  const _Float16* rp = act + row * TS;
  v16h a;
#pragma unroll
  for (int v = 0; v < 8; ++v) {
    int kk = k0 + ((v < 4) ? 0 : 16) + hs * 8 + (v & 3) * 2;
    a[2 * v]     = rp[kk];
    a[2 * v + 1] = rp[kk + 1];
  }
  return a;
}

// all 8 B fragments (4 N-tiles x 2 K-steps) of a 64x64 weight matrix,
// converted f32->f16 and held in VGPRs for the lifetime of the wave
struct BFrags { v16h f[8]; };

__device__ __forceinline__ v16h load_bfrag(const float* W, int nt, int k0,
                                           int lane) {
  int n  = nt * 16 + (lane & 15);
  int hs = lane >> 4;
  v16h b;
#pragma unroll
  for (int v = 0; v < 8; ++v) {
    int kk = k0 + ((v < 4) ? 0 : 16) + hs * 8 + (v & 3) * 2;
    b[2 * v]     = (_Float16)W[kk * 64 + n];
    b[2 * v + 1] = (_Float16)W[(kk + 1) * 64 + n];
  }
  return b;
}

__device__ __forceinline__ BFrags load_bfrags(const float* W, int lane) {
  BFrags B;
#pragma unroll
  for (int nt = 0; nt < 4; ++nt) {
    B.f[nt * 2 + 0] = load_bfrag(W, nt, 0, lane);
    B.f[nt * 2 + 1] = load_bfrag(W, nt, 32, lane);
  }
  return B;
}

// 32x64 = (32x64 f16 act tile) @ (register-resident 64x64 f16 weights) + bias
__device__ __forceinline__ void wmma_mm64(const _Float16* act, const BFrags& B,
                                          const float* bias, float* outT,
                                          int lane) {
  int ln = lane & 15, hs = lane >> 4;
#pragma unroll
  for (int nt = 0; nt < 4; ++nt) {
    int col  = nt * 16 + ln;
    float bi = bias ? bias[col] : 0.0f;
#pragma unroll
    for (int mt = 0; mt < 2; ++mt) {
      v16h a0 = load_afrag(act, mt, 0, lane);
      v16h a1 = load_afrag(act, mt, 32, lane);
      v8f acc = {};
      acc = __builtin_amdgcn_wmma_f32_16x16x32_f16(false, a0, false, B.f[nt * 2],
                                                   (short)0, acc, false, false);
      acc = __builtin_amdgcn_wmma_f32_16x16x32_f16(false, a1, false, B.f[nt * 2 + 1],
                                                   (short)0, acc, false, false);
#pragma unroll
      for (int e = 0; e < 8; ++e) {
        int row = mt * 16 + hs * 8 + e;
        outT[row * TS + col] = acc[e] + bi;
      }
    }
  }
}

// ---- main staged kernel ---------------------------------------------------
// STAGE 1: stats(x1)            STAGE 2: stats(v - geom)
// STAGE 3: stats(q)             STAGE 4: stats(a1pre)
// STAGE 5: full forward -> out
template <int STAGE>
__global__ __launch_bounds__(64) void k_stage(Args A) {
  __shared__ float    sW1[192];
  __shared__ float    sPar[512];
  __shared__ float    sDB2[64];
  __shared__ float    sGB[64];
  __shared__ float    sAW2[512];
  __shared__ float    sAB2[8];
  __shared__ _Float16 tA[2][32 * TS];
  __shared__ float    tG[2][32 * TS];
  __shared__ float    tB[2][32 * TS];
  __shared__ float    sABuf[2][32 * 8];

  int tid = threadIdx.x;
  for (int i = tid; i < 192; i += 64) sW1[i] = A.dW1[i];
  for (int i = tid; i < 512; i += 64) sPar[i] = A.params[i];
  if (tid < 64) { sDB2[tid] = A.dB2[tid]; sGB[tid] = A.gB[tid]; }
  for (int i = tid; i < 512; i += 64) sAW2[i] = A.aW2[i];
  if (tid < 8) sAB2[tid] = A.aB2[tid];
  __syncthreads();

  int w = tid >> 5, lane = tid & 31;
  _Float16* tA_ = tA[w];
  float* tG_ = tG[w];
  float* tB_ = tB[w];
  float* aB_ = sABuf[w];
  const float* sc1 = sPar + 0;   const float* sh1 = sPar + 64;
  const float* sc2 = sPar + 128; const float* sh2 = sPar + 192;
  const float* sc3 = sPar + 256; const float* sh3 = sPar + 320;
  const float* sc4 = sPar + 384; const float* sh4 = sPar + 448;

  // loop-invariant register-resident weight fragments
  BFrags Bd, Bg, Ba;
  if (STAGE >= 2) Bd = load_bfrags(A.dW2, lane);
  if (STAGE >= 3) Bg = load_bfrags(A.gW, lane);
  if (STAGE >= 4) Ba = load_bfrags(A.aW1, lane);

  float s0 = 0.f, q0 = 0.f, s1 = 0.f, q1 = 0.f;
  int c0 = lane, c1 = lane + 32;
  int gw = blockIdx.x * 2 + w;
  int nw = gridDim.x * 2;

  for (int m = gw; m < M_PTS; m += nw) {
    // lane <-> neighbor h
    int idx = A.nbi[m * H_NB + lane];
    float qx = A.q_pts[m * 3 + 0], qy = A.q_pts[m * 3 + 1], qz = A.q_pts[m * 3 + 2];
    float d0 = A.s_pts[idx * 3 + 0] - qx;
    float d1 = A.s_pts[idx * 3 + 1] - qy;
    float d2 = A.s_pts[idx * 3 + 2] - qz;

    if (STAGE == 1) {                      // raw x1 = nbrs @ delta_W1
      for (int c = 0; c < 64; ++c)
        tG_[lane * TS + c] = d0 * sW1[c] + d1 * sW1[64 + c] + d2 * sW1[128 + c];
      LDS_FENCE();
      for (int h = 0; h < 32; ++h) {
        float a = tG_[h * TS + c0]; s0 += a; q0 += a * a;
        float b = tG_[h * TS + c1]; s1 += b; q1 += b * b;
      }
      continue;
    }

    // act1 = lrelu(bn1(x1))  (f16 tile)
    for (int c = 0; c < 64; ++c) {
      float x = d0 * sW1[c] + d1 * sW1[64 + c] + d2 * sW1[128 + c];
      x = x * sc1[c] + sh1[c];
      x = x >= 0.f ? x : SLOPE * x;
      tA_[lane * TS + c] = (_Float16)x;
    }
    LDS_FENCE();
    wmma_mm64(tA_, Bd, sDB2, tG_, lane);         // geom -> tG
    LDS_FENCE();

    const float* vr = A.s_feats + (size_t)idx * 64;   // L2-resident gather
    if (STAGE == 2) {                     // vmg = v - geom (in place)
      for (int c = 0; c < 64; ++c)
        tG_[lane * TS + c] = vr[c] - tG_[lane * TS + c];
      LDS_FENCE();
      for (int h = 0; h < 32; ++h) {
        float a = tG_[h * TS + c0]; s0 += a; q0 += a * a;
        float b = tG_[h * TS + c1]; s1 += b; q1 += b * b;
      }
      continue;
    }

    // act2 = lrelu(bn2(v - geom))
    for (int c = 0; c < 64; ++c) {
      float x = vr[c] - tG_[lane * TS + c];
      x = x * sc2[c] + sh2[c];
      x = x >= 0.f ? x : SLOPE * x;
      tA_[lane * TS + c] = (_Float16)x;
    }
    LDS_FENCE();
    wmma_mm64(tA_, Bg, sGB, tB_, lane);          // v2 -> tB
    LDS_FENCE();

    if (STAGE == 3) {                     // q = v2[h=0] - geom
      for (int h = 0; h < 32; ++h) {
        float a = tB_[c0] - tG_[h * TS + c0]; s0 += a; q0 += a * a;
        float b = tB_[c1] - tG_[h * TS + c1]; s1 += b; q1 += b * b;
      }
      continue;
    }

    // act3 = lrelu(bn3(q))
    for (int c = 0; c < 64; ++c) {
      float x = tB_[c] - tG_[lane * TS + c];
      x = x * sc3[c] + sh3[c];
      x = x >= 0.f ? x : SLOPE * x;
      tA_[lane * TS + c] = (_Float16)x;
    }
    LDS_FENCE();
    wmma_mm64(tA_, Ba, nullptr, tG_, lane);      // a1pre -> tG
    LDS_FENCE();

    if (STAGE == 4) {
      for (int h = 0; h < 32; ++h) {
        float a = tG_[h * TS + c0]; s0 += a; q0 += a * a;
        float b = tG_[h * TS + c1]; s1 += b; q1 += b * b;
      }
      continue;
    }

    // act4 = lrelu(bn4(a1pre))
    for (int c = 0; c < 64; ++c) {
      float x = tG_[lane * TS + c] * sc4[c] + sh4[c];
      x = x >= 0.f ? x : SLOPE * x;
      tA_[lane * TS + c] = (_Float16)x;
    }
    LDS_FENCE();

    // logits = act4 @ alpha_W2 + b2   (lane = row h, 8 cols)
    float lg[8];
#pragma unroll
    for (int j = 0; j < 8; ++j) lg[j] = sAB2[j];
    for (int k = 0; k < 64; ++k) {
      float av = (float)tA_[lane * TS + k];
#pragma unroll
      for (int j = 0; j < 8; ++j) lg[j] += av * sAW2[k * 8 + j];
    }
    // softmax over neighbors (cross-lane, wave32)
#pragma unroll
    for (int j = 0; j < 8; ++j) {
      float mx = lg[j];
      for (int off = 16; off >= 1; off >>= 1) mx = fmaxf(mx, __shfl_xor(mx, off, 32));
      float e = __expf(lg[j] - mx);
      float sm = e;
      for (int off = 16; off >= 1; off >>= 1) sm += __shfl_xor(sm, off, 32);
      aB_[lane * 8 + j] = e / sm;
    }
    LDS_FENCE();
    // grouped weighted sum over neighbors
    float o0 = 0.f, o1 = 0.f;
    for (int h = 0; h < 32; ++h) {
      o0 += tB_[h * TS + c0] * aB_[h * 8 + (c0 >> 3)];
      o1 += tB_[h * TS + c1] * aB_[h * 8 + (c1 >> 3)];
    }
    A.out[(size_t)m * 64 + c0] = o0;
    A.out[(size_t)m * 64 + c1] = o1;
  }

  if (STAGE >= 1 && STAGE <= 4) {
    float* p = A.part + (size_t)(STAGE - 1) * (NREP * 128) +
               (size_t)(gw & (NREP - 1)) * 128;
    atomicAdd(p + c0, s0);       atomicAdd(p + 64 + c0, q0);
    atomicAdd(p + c1, s1);       atomicAdd(p + 64 + c1, q1);
  }
}

// fold BN stats into per-channel scale/shift
__global__ void k_fin(const float* part, float* params, const float* g,
                      const float* b, float invNT) {
  int c = threadIdx.x;
  if (c >= 64) return;
  float s = 0.f, sq = 0.f;
  for (int r = 0; r < NREP; ++r) {
    s  += part[r * 128 + c];
    sq += part[r * 128 + 64 + c];
  }
  float mean = s * invNT;
  float var  = sq * invNT - mean * mean;
  float rstd = rsqrtf(var + 1e-5f);
  float sc   = g[c] * rstd;
  params[c]      = sc;
  params[64 + c] = b[c] - mean * sc;
}

extern "C" void kernel_launch(void* const* d_in, const int* in_sizes, int n_in,
                              void* d_out, int out_size, void* d_ws, size_t ws_size,
                              hipStream_t stream) {
  float* ws = (float*)d_ws;
  const size_t PART_FLOATS = 4 * NREP * 128;  // 32768

  Args A;
  A.q_pts   = (const float*)d_in[0];
  A.s_pts   = (const float*)d_in[1];
  A.s_feats = (const float*)d_in[2];
  A.nbi     = (const int*)  d_in[3];
  A.dW1     = (const float*)d_in[4];
  A.dW2     = (const float*)d_in[7];
  A.dB2     = (const float*)d_in[8];
  A.gW      = (const float*)d_in[11];
  A.gB      = (const float*)d_in[12];
  A.aW1     = (const float*)d_in[15];
  A.aW2     = (const float*)d_in[18];
  A.aB2     = (const float*)d_in[19];
  A.part    = ws;
  A.params  = ws + PART_FLOATS;
  A.out     = (float*)d_out;

  hipMemsetAsync(d_ws, 0, PART_FLOATS * sizeof(float), stream);

  const float invNT = 1.0f / (float)(M_PTS * H_NB);
  dim3 blk(64), grd(1024);
  float* params = ws + PART_FLOATS;

  k_stage<1><<<grd, blk, 0, stream>>>(A);
  k_fin<<<1, 64, 0, stream>>>(ws + 0 * NREP * 128, params + 0,
                              (const float*)d_in[5],  (const float*)d_in[6],  invNT);
  k_stage<2><<<grd, blk, 0, stream>>>(A);
  k_fin<<<1, 64, 0, stream>>>(ws + 1 * NREP * 128, params + 128,
                              (const float*)d_in[9],  (const float*)d_in[10], invNT);
  k_stage<3><<<grd, blk, 0, stream>>>(A);
  k_fin<<<1, 64, 0, stream>>>(ws + 2 * NREP * 128, params + 256,
                              (const float*)d_in[13], (const float*)d_in[14], invNT);
  k_stage<4><<<grd, blk, 0, stream>>>(A);
  k_fin<<<1, 64, 0, stream>>>(ws + 3 * NREP * 128, params + 384,
                              (const float*)d_in[16], (const float*)d_in[17], invNT);
  k_stage<5><<<grd, blk, 0, stream>>>(A);
}
